// _Seq2Seq_9302899163732
// MI455X (gfx1250) — matrix operand order repaired and checked by reading, not compiled, not run on hardware
//
#include <hip/hip_runtime.h>
#include <hip/hip_bf16.h>

// ---------------------------------------------------------------------------
// Seq2Seq LSTM encoder/decoder for MI455X (gfx1250, wave32, WMMA bf16).
//   B=128, S=1024, IN=256, H=512, OUT=513 (=512+1), MAX_LEN=100
//   * bf16 V_WMMA_F32_16X16X32_BF16 GEMMs, fp32 accumulators
//   * x pre-converted to bf16 once (ws permitting); hw v_cvt_pk_bf16_f32
//   * templated step kernel -> fully unrolled K loops
//   * fused register-resident LSTM cell; double-buffered h (bf16), c (f32)
//   * fused projection + log_softmax (LDS + wave32 shfl reductions)
// ---------------------------------------------------------------------------

typedef __bf16 bf16_t;
typedef __attribute__((ext_vector_type(16))) __bf16 v16bf;
typedef __attribute__((ext_vector_type(8)))  __bf16 v8bf;
typedef __attribute__((ext_vector_type(2)))  __bf16 v2bf;
typedef __attribute__((ext_vector_type(8)))  float  v8f;

#define HID   512
#define GATES 2048
#define BATCH 128
#define SEQ   1024
#define INDIM 256
#define OUTD  513
#define NPAD  640       // 40 tiles of 16: uniform 5 tiles/wave in proj kernel
#define MAXL  100

// ---- f32 -> bf16 (RNE). Prefer hardware pack op. ---------------------------
#if __has_builtin(__builtin_amdgcn_cvt_pk_bf16_f32)
#define HAVE_CVT_PK_BF16 1
#else
#define HAVE_CVT_PK_BF16 0
#endif

__device__ __forceinline__ bf16_t f2bf(float f) {
#if HAVE_CVT_PK_BF16
  v2bf p = __builtin_amdgcn_cvt_pk_bf16_f32(f, 0.0f);
  return p[0];
#else
  unsigned u = __builtin_bit_cast(unsigned, f);
  unsigned r = (u + 0x7FFFu + ((u >> 16) & 1u)) >> 16;
  return __builtin_bit_cast(bf16_t, (unsigned short)r);
#endif
}

// convert a pair into vector slots i, i+1
__device__ __forceinline__ void cvt2(v16bf& a, int i, float x, float y) {
#if HAVE_CVT_PK_BF16
  v2bf p = __builtin_amdgcn_cvt_pk_bf16_f32(x, y);
  a[i] = p[0]; a[i + 1] = p[1];
#else
  a[i] = f2bf(x); a[i + 1] = f2bf(y);
#endif
}

__device__ __forceinline__ float sigmf(float x) {
  return 1.0f / (1.0f + __expf(-x));
}
__device__ __forceinline__ float tanh_f(float x) {
  x = fminf(fmaxf(x, -15.0f), 15.0f);
  float e = __expf(-2.0f * x);
  return (1.0f - e) / (1.0f + e);
}

// ---- WMMA fragment loaders (contiguous k = 16*(lane/16) + s in A and B, wave32) ----------

// A fragment 16x32 bf16 from fp32 source (inline convert, fallback path).
__device__ __forceinline__ v16bf load_a_f32(const float* __restrict__ base,
                                            long rstride, int m0, int k0, int lane) {
  int m  = m0 + (lane & 15);
  int kb = k0 + ((lane >> 4) << 4);
  const float* row = base + (long)m * rstride + kb;
  float4 f0 = *(const float4*)(row);
  float4 f1 = *(const float4*)(row + 4);
  float4 f2 = *(const float4*)(row + 8);
  float4 f3 = *(const float4*)(row + 12);
  v16bf a;
  cvt2(a, 0,  f0.x, f0.y); cvt2(a, 2,  f0.z, f0.w);
  cvt2(a, 4,  f1.x, f1.y); cvt2(a, 6,  f1.z, f1.w);
  cvt2(a, 8,  f2.x, f2.y); cvt2(a, 10, f2.z, f2.w);
  cvt2(a, 12, f3.x, f3.y); cvt2(a, 14, f3.z, f3.w);
  return a;
}

// A fragment 16x32 bf16 from bf16 source (two contiguous 16B loads).
__device__ __forceinline__ v16bf load_a_bf(const bf16_t* __restrict__ base,
                                           long rstride, int m0, int k0, int lane) {
  int m  = m0 + (lane & 15);
  int kb = k0 + ((lane >> 4) << 4);
  const bf16_t* row = base + (long)m * rstride + kb;
  v8bf c0 = *(const v8bf*)(row);
  v8bf c1 = *(const v8bf*)(row + 8);
  v16bf a;
#pragma unroll
  for (int i = 0; i < 8; ++i) { a[i] = c0[i]; a[8 + i] = c1[i]; }
  return a;
}

// B fragment 32x16 bf16 from row-major weight W[N,K]; one 32B load per lane.
__device__ __forceinline__ v16bf load_b(const bf16_t* __restrict__ W, long ldw,
                                        int n0, int k0, int lane) {
  int n  = n0 + (lane & 15);
  int ko = k0 + ((lane >> 4) << 4);
  return *(const v16bf*)(W + (long)n * ldw + ko);
}

__device__ __forceinline__ v8f wmma_bf16(v16bf a, v16bf b, v8f c) {
  return __builtin_amdgcn_wmma_f32_16x16x32_bf16(false, a, false, b,
                                                 (short)0, c, false, false);
}

// ---------------------------------------------------------------------------
// Fused LSTM step (templated: KX = input width, XF32 = fp32 input path).
// grid = (8 mTiles, 8 hidden-slices of 64), block = 128 (4 waves).
// Wave w of slice s owns hidden tile j0=(s*4+w)*16 with matching i/f/g/o
// accumulators -> cell update entirely in registers. h double-buffered.
// ---------------------------------------------------------------------------
template <int KX, bool XF32>
__global__ __launch_bounds__(128) void lstm_step_kernel(
    const float*  __restrict__ xf,                      // fp32 input (XF32)
    const bf16_t* __restrict__ xb,                      // bf16 input (!XF32)
    long x_stride,                                      // input row stride
    const bf16_t* __restrict__ Wih,                     // [GATES, KX] bf16
    const bf16_t* __restrict__ Whh,                     // [GATES, HID] bf16
    const float*  __restrict__ bias,                    // [GATES] (bih+bhh)
    const bf16_t* __restrict__ h_in,                    // [BATCH, HID] bf16
    float*        __restrict__ cst,                     // [BATCH, HID] f32
    bf16_t*       __restrict__ h_out) {                 // [BATCH, HID] bf16
  const int lane = threadIdx.x & 31;
  const int wave = threadIdx.x >> 5;
  const int m0   = blockIdx.x * 16;
  const int j0   = (blockIdx.y * 4 + wave) * 16;

  v8f acc[4] = {};   // i, f, g, o tiles (16x16 f32 each)

  // ---- input-projection part -------------------------------------------
#pragma unroll
  for (int k0 = 0; k0 < KX; k0 += 32) {
    v16bf a;
    if constexpr (XF32) a = load_a_f32(xf, x_stride, m0, k0, lane);
    else                a = load_a_bf (xb, x_stride, m0, k0, lane);
#pragma unroll
    for (int g = 0; g < 4; ++g) {
      v16bf b = load_b(Wih, KX, g * HID + j0, k0, lane);
      acc[g] = wmma_bf16(a, b, acc[g]);
    }
  }

  // ---- recurrent part ---------------------------------------------------
#pragma unroll
  for (int k0 = 0; k0 < HID; k0 += 32) {
    v16bf a = load_a_bf(h_in, HID, m0, k0, lane);
#pragma unroll
    for (int g = 0; g < 4; ++g) {
      v16bf b = load_b(Whh, HID, g * HID + j0, k0, lane);
      acc[g] = wmma_bf16(a, b, acc[g]);
    }
  }

  // ---- fused LSTM cell --------------------------------------------------
  const int j = j0 + (lane & 15);
  const float bi = bias[0 * HID + j];
  const float bf = bias[1 * HID + j];
  const float bg = bias[2 * HID + j];
  const float bo = bias[3 * HID + j];
  const int mrow0 = m0 + ((lane >> 4) << 3);
#pragma unroll
  for (int r = 0; r < 8; ++r) {
    long idx = (long)(mrow0 + r) * HID + j;
    float iv = sigmf (acc[0][r] + bi);
    float fv = sigmf (acc[1][r] + bf);
    float gv = tanh_f(acc[2][r] + bg);
    float ov = sigmf (acc[3][r] + bo);
    float cn = fv * cst[idx] + iv * gv;
    cst[idx]   = cn;
    h_out[idx] = f2bf(ov * tanh_f(cn));
  }
}

// ---------------------------------------------------------------------------
// Fused projection + log_softmax. grid = 8, block = 256 (8 waves),
// N padded to 640 -> exactly 5 N-tiles per wave (constant unroll).
// ---------------------------------------------------------------------------
__global__ __launch_bounds__(256) void proj_kernel(
    const bf16_t* __restrict__ h,      // [BATCH, HID] bf16
    const bf16_t* __restrict__ Wo,     // [NPAD, HID] bf16 (rows >=513 zero)
    const float*  __restrict__ ob,     // [OUTD] f32
    float*        __restrict__ out,    // [BATCH, MAXL, OUTD] f32
    int t) {
  __shared__ float ls[16 * NPAD];
  const int lane = threadIdx.x & 31;
  const int wave = threadIdx.x >> 5;
  const int m0   = blockIdx.x * 16;

  v8f acc[5] = {};
#pragma unroll
  for (int k0 = 0; k0 < HID; k0 += 32) {
    v16bf a = load_a_bf(h, HID, m0, k0, lane);
#pragma unroll
    for (int u = 0; u < 5; ++u) {
      int n0 = (wave + 8 * u) * 16;
      v16bf b = load_b(Wo, HID, n0, k0, lane);
      acc[u] = wmma_bf16(a, b, acc[u]);
    }
  }

  const int r0 = (lane >> 4) << 3;
#pragma unroll
  for (int u = 0; u < 5; ++u) {
    int col = (wave + 8 * u) * 16 + (lane & 15);
    float bv = (col < OUTD) ? ob[col] : 0.0f;
#pragma unroll
    for (int r = 0; r < 8; ++r)
      ls[(r0 + r) * NPAD + col] = acc[u][r] + bv;
  }
  __syncthreads();

#pragma unroll
  for (int rr = 0; rr < 2; ++rr) {
    int row = wave * 2 + rr;
    const float* lr = &ls[row * NPAD];
    float mx = -__builtin_inff();
    for (int jj = lane; jj < OUTD; jj += 32) mx = fmaxf(mx, lr[jj]);
#pragma unroll
    for (int off = 16; off > 0; off >>= 1) mx = fmaxf(mx, __shfl_xor(mx, off, 32));
    float sum = 0.0f;
    for (int jj = lane; jj < OUTD; jj += 32) sum += __expf(lr[jj] - mx);
#pragma unroll
    for (int off = 16; off > 0; off >>= 1) sum += __shfl_xor(sum, off, 32);
    float lse = mx + __logf(sum);
    float* orow = out + ((long)(m0 + row) * MAXL + t) * OUTD;
    for (int jj = lane; jj < OUTD; jj += 32) orow[jj] = lr[jj] - lse;
  }
}

// ------------------------------ prep kernels --------------------------------
__global__ void cvt_f32_bf16(bf16_t* __restrict__ dst,
                             const float* __restrict__ src, int n) {
  int i = blockIdx.x * blockDim.x + threadIdx.x;
  if (i < n) dst[i] = f2bf(src[i]);
}
// vectorized x conversion: 8 floats -> 8 bf16 per thread
__global__ void cvt_x8(bf16_t* __restrict__ dst,
                       const float* __restrict__ src, long n8) {
  long i = (long)blockIdx.x * blockDim.x + threadIdx.x;
  if (i < n8) {
    const float4* s = (const float4*)src + 2 * i;
    float4 f0 = s[0], f1 = s[1];
    v16bf tmp;
    cvt2(tmp, 0, f0.x, f0.y); cvt2(tmp, 2, f0.z, f0.w);
    cvt2(tmp, 4, f1.x, f1.y); cvt2(tmp, 6, f1.z, f1.w);
    v8bf o;
#pragma unroll
    for (int q = 0; q < 8; ++q) o[q] = tmp[q];
    *((v8bf*)dst + i) = o;
  }
}
__global__ void pad_outw(bf16_t* __restrict__ dst,
                         const float* __restrict__ src) {     // [NPAD,512]
  int i = blockIdx.x * blockDim.x + threadIdx.x;
  if (i < NPAD * HID) {
    int row = i >> 9, col = i & 511;
    dst[i] = (row < OUTD) ? f2bf(src[row * HID + col]) : f2bf(0.0f);
  }
}
__global__ void add_bias(float* __restrict__ d, const float* __restrict__ a,
                         const float* __restrict__ b, int n) {
  int i = blockIdx.x * blockDim.x + threadIdx.x;
  if (i < n) d[i] = a[i] + b[i];
}
__global__ void zero16(uint4* __restrict__ p, int n16) {
  int i = blockIdx.x * blockDim.x + threadIdx.x;
  if (i < n16) p[i] = make_uint4(0u, 0u, 0u, 0u);
}

// ------------------------------ workspace map -------------------------------
enum : size_t {
  OFF_EWIH = 0,
  OFF_EWHH = OFF_EWIH + (size_t)GATES * INDIM * 2,
  OFF_DWIH = OFF_EWHH + (size_t)GATES * HID * 2,
  OFF_DWHH = OFF_DWIH + (size_t)GATES * HID * 2,
  OFF_OW   = OFF_DWHH + (size_t)GATES * HID * 2,
  OFF_EB   = OFF_OW   + (size_t)NPAD * HID * 2,
  OFF_DB   = OFF_EB   + (size_t)GATES * 4,
  OFF_HA   = OFF_DB   + (size_t)GATES * 4,
  OFF_HB   = OFF_HA   + (size_t)BATCH * HID * 2,
  OFF_Z    = OFF_HB   + (size_t)BATCH * HID * 2,
  OFF_C    = OFF_Z    + (size_t)BATCH * HID * 2,
  OFF_END  = OFF_C    + (size_t)BATCH * HID * 4,      // ~8.7 MB baseline
  OFF_XBF  = (OFF_END + 255) & ~(size_t)255,
  OFF_FULL = OFF_XBF  + (size_t)BATCH * SEQ * INDIM * 2,  // +64 MB optional
};

extern "C" void kernel_launch(void* const* d_in, const int* in_sizes, int n_in,
                              void* d_out, int out_size, void* d_ws, size_t ws_size,
                              hipStream_t stream) {
  (void)in_sizes; (void)n_in; (void)out_size;
  const float* x       = (const float*)d_in[0];
  const float* enc_Wih = (const float*)d_in[1];
  const float* enc_Whh = (const float*)d_in[2];
  const float* enc_bih = (const float*)d_in[3];
  const float* enc_bhh = (const float*)d_in[4];
  const float* dec_Wih = (const float*)d_in[5];
  const float* dec_Whh = (const float*)d_in[6];
  const float* dec_bih = (const float*)d_in[7];
  const float* dec_bhh = (const float*)d_in[8];
  const float* out_W   = (const float*)d_in[9];
  const float* out_b   = (const float*)d_in[10];
  float* out = (float*)d_out;

  char* ws = (char*)d_ws;
  bf16_t* eWih = (bf16_t*)(ws + OFF_EWIH);
  bf16_t* eWhh = (bf16_t*)(ws + OFF_EWHH);
  bf16_t* dWih = (bf16_t*)(ws + OFF_DWIH);
  bf16_t* dWhh = (bf16_t*)(ws + OFF_DWHH);
  bf16_t* oW   = (bf16_t*)(ws + OFF_OW);
  float*  eb   = (float*)(ws + OFF_EB);
  float*  db   = (float*)(ws + OFF_DB);
  bf16_t* hA   = (bf16_t*)(ws + OFF_HA);
  bf16_t* hB   = (bf16_t*)(ws + OFF_HB);
  bf16_t* zbuf = (bf16_t*)(ws + OFF_Z);
  float*  cbuf = (float*)(ws + OFF_C);
  bf16_t* xbf  = (bf16_t*)(ws + OFF_XBF);

  const bool use_xbf = (ws_size >= OFF_FULL);   // deterministic across calls

  // ---- prep --------------------------------------------------------------
  auto blks = [](long n, int t) { return (int)((n + t - 1) / t); };
  cvt_f32_bf16<<<blks(GATES * INDIM, 256), 256, 0, stream>>>(eWih, enc_Wih, GATES * INDIM);
  cvt_f32_bf16<<<blks(GATES * HID,   256), 256, 0, stream>>>(eWhh, enc_Whh, GATES * HID);
  cvt_f32_bf16<<<blks(GATES * HID,   256), 256, 0, stream>>>(dWih, dec_Wih, GATES * HID);
  cvt_f32_bf16<<<blks(GATES * HID,   256), 256, 0, stream>>>(dWhh, dec_Whh, GATES * HID);
  pad_outw    <<<blks(NPAD * HID,    256), 256, 0, stream>>>(oW, out_W);
  add_bias    <<<blks(GATES, 256),         256, 0, stream>>>(eb, enc_bih, enc_bhh, GATES);
  add_bias    <<<blks(GATES, 256),         256, 0, stream>>>(db, dec_bih, dec_bhh, GATES);
  const int zero_n16 = (int)((OFF_END - OFF_HA) / 16);   // hA, hB, zbuf, c
  zero16      <<<blks(zero_n16, 256),      256, 0, stream>>>((uint4*)(ws + OFF_HA), zero_n16);

  const dim3 sgrid(8, 8);   // 8 m-tiles x 8 hidden slices

  // ---- encoder: 1024 steps, h ping-pong (hA zeroed start) ----------------
  if (use_xbf) {
    const long n8 = (long)BATCH * SEQ * INDIM / 8;
    cvt_x8<<<blks(n8, 256), 256, 0, stream>>>(xbf, x, n8);
    for (int t = 0; t < SEQ; ++t) {
      bf16_t* hin  = (t & 1) ? hB : hA;
      bf16_t* hout = (t & 1) ? hA : hB;
      lstm_step_kernel<INDIM, false><<<sgrid, 128, 0, stream>>>(
          nullptr, xbf + (long)t * INDIM, (long)SEQ * INDIM,
          eWih, eWhh, eb, hin, cbuf, hout);
    }
  } else {
    for (int t = 0; t < SEQ; ++t) {
      bf16_t* hin  = (t & 1) ? hB : hA;
      bf16_t* hout = (t & 1) ? hA : hB;
      lstm_step_kernel<INDIM, true><<<sgrid, 128, 0, stream>>>(
          x + (long)t * INDIM, nullptr, (long)SEQ * INDIM,
          eWih, eWhh, eb, hin, cbuf, hout);
    }
  }
  // SEQ even -> final encoder h lands in hA

  // ---- decoder: 100 steps; input = prev h (zeros at t=0) -----------------
  bf16_t* cur = hA;
  bf16_t* oth = hB;
  for (int t = 0; t < MAXL; ++t) {
    const bf16_t* xin = (t == 0) ? zbuf : cur;
    lstm_step_kernel<HID, false><<<sgrid, 128, 0, stream>>>(
        nullptr, xin, (long)HID, dWih, dWhh, db, cur, cbuf, oth);
    proj_kernel<<<8, 256, 0, stream>>>(oth, oW, out_b, out, t);
    bf16_t* tmp = cur; cur = oth; oth = tmp;
  }
}
